// Head_17746804867127
// MI455X (gfx1250) — matrix operand order repaired and checked by reading, not compiled, not run on hardware
//
#include <hip/hip_runtime.h>

#define T_SEQ 2048
#define NB    32
#define CDIM  1024
#define HDIM  64

typedef __attribute__((ext_vector_type(16))) __bf16         v16bf;
typedef __attribute__((ext_vector_type(8)))  __bf16         v8bf;
typedef __attribute__((ext_vector_type(4)))  __bf16         v4bf;
typedef __attribute__((ext_vector_type(8)))  float          v8f;
typedef __attribute__((ext_vector_type(4)))  float          v4f;
typedef __attribute__((ext_vector_type(8)))  unsigned short u16x8;

union V16U {
    v16bf v;
    u16x8 h[2];
    v4bf  q[4];
    unsigned short s[16];
};

// native float -> bf16 convert (backend emits v_cvt bf16 ops, RNE)
__device__ __forceinline__ unsigned short f2bf(float f) {
    return __builtin_bit_cast(unsigned short, (__bf16)f);
}

__device__ __forceinline__ v8f wmma_bf16(v16bf a, v16bf b, v8f c) {
    return __builtin_amdgcn_wmma_f32_16x16x32_bf16(
        /*neg_a=*/false, a, /*neg_b=*/false, b,
        /*c_mod=*/(short)0, c, /*reuse_a=*/false, /*reuse_b=*/false);
}

// A fragment (16x32 bf16), row-major source, tile origin `base`, stride in elements.
// A layout: row M = lane&15, K(e) = 16*(e>>3) + 8*(lane>>4) + (e&7)  -> two 16B runs.
__device__ __forceinline__ v16bf load_a_rm(const unsigned short* base, int stride, int lane) {
    int m = lane & 15, hi = lane >> 4;
    V16U r;
    r.h[0] = *(const u16x8*)(base + (long)m * stride + 8 * hi);
    r.h[1] = *(const u16x8*)(base + (long)m * stride + 16 + 8 * hi);
    return r.v;
}

// B fragment (32x16 bf16) where element B[kk][n] = base[n*stride + kk] (K contiguous).
// B layout: col N = lane&15, K(e) = 16*(e>>3) + 8*(lane>>4) + (e&7)  -> two 16B runs.
__device__ __forceinline__ v16bf load_b_kc(const unsigned short* base, int stride, int lane) {
    int n = lane & 15, hi = lane >> 4;
    const unsigned short* p = base + (long)n * stride + 8 * hi;
    V16U r;
    r.h[0] = *(const u16x8*)(p);
    r.h[1] = *(const u16x8*)(p + 16);
    return r.v;
}

// -------- Kernel 0: convert W[1024][64] fp32 -> WT[64][1024] bf16 (x3 matrices) ------
__global__ void __launch_bounds__(256)
cvt_w(const float* __restrict__ Wk, const float* __restrict__ Wq,
      const float* __restrict__ Wv, unsigned short* __restrict__ wt) {
    int idx = blockIdx.x * 256 + threadIdx.x;       // 0 .. 3*64*1024-1
    int w = idx >> 16;                               // matrix id
    int r = idx & 65535;                             // h*1024 + c
    int h = r >> 10, c = r & 1023;
    const float* W = (w == 0) ? Wk : ((w == 1) ? Wq : Wv);
    wt[idx] = f2bf(W[c * HDIM + h]);
}

// ---------------- Kernel 1: k,q,v projections (x read exactly once) ----------------
__global__ void __launch_bounds__(256)
proj_kqv(const float* __restrict__ x, const unsigned short* __restrict__ wt,
         unsigned short* __restrict__ kb, unsigned short* __restrict__ qb,
         unsigned short* __restrict__ vt) {
    int wid  = (blockIdx.x * blockDim.x + threadIdx.x) >> 5;   // tile id 0..4095
    int lane = threadIdx.x & 31;
    int m = lane & 15, hi = lane >> 4;
    long g0 = (long)wid * 16;                                   // first global row of tile
    const float* xr = x + (g0 + m) * CDIM;                      // this lane's x row

    const v8f vzero = {};
    v8f acc[3][4];
    #pragma unroll
    for (int w = 0; w < 3; ++w)
        #pragma unroll
        for (int h = 0; h < 4; ++h) acc[w][h] = vzero;

    for (int c0 = 0; c0 < CDIM; c0 += 32) {
        // A fragment from x: two 32B runs per lane, packed fp32->bf16 converts
        const float* p0 = xr + c0 + 8 * hi;        // e = 0..7
        const float* p1 = xr + c0 + 16 + 8 * hi;   // e = 8..15
        V16U a;
        a.q[0] = __builtin_convertvector(*(const v4f*)(p0),     v4bf);
        a.q[1] = __builtin_convertvector(*(const v4f*)(p0 + 4), v4bf);
        a.q[2] = __builtin_convertvector(*(const v4f*)(p1),     v4bf);
        a.q[3] = __builtin_convertvector(*(const v4f*)(p1 + 4), v4bf);
        #pragma unroll
        for (int w = 0; w < 3; ++w) {
            #pragma unroll
            for (int h = 0; h < 4; ++h) {
                // B[kk][n] = W[c0+kk][h*16+n] = WT[w][h*16+n][c0+kk] -> contiguous
                v16bf b = load_b_kc(wt + (long)w * HDIM * CDIM + (long)(h * 16) * CDIM + c0,
                                    CDIM, lane);
                acc[w][h] = wmma_bf16(a.v, b, acc[w][h]);
            }
        }
    }

    // k, q: row-major bf16 [B*T][64]  (C-layout scatter: row = r+8*hi, col = h*16+m)
    #pragma unroll
    for (int h = 0; h < 4; ++h)
        #pragma unroll
        for (int r = 0; r < 8; ++r) {
            long row = g0 + r + 8 * hi;
            kb[row * HDIM + h * 16 + m] = f2bf(acc[0][h][r]);
            qb[row * HDIM + h * 16 + m] = f2bf(acc[1][h][r]);
        }
    // v transposed: vt[b][h][t]; the 8 accumulator rows are 8 consecutive t -> packed b128
    int batch = (int)(g0 >> 11);
    int t0    = (int)(g0 & (T_SEQ - 1));
    #pragma unroll
    for (int h = 0; h < 4; ++h) {
        u16x8 pk;
        #pragma unroll
        for (int r = 0; r < 8; ++r) pk[r] = f2bf(acc[2][h][r]);
        long off = ((long)batch * HDIM + h * 16 + m) * T_SEQ + t0 + 8 * hi;
        *(u16x8*)(vt + off) = pk;
    }
}

// ---------------- Kernel 2: flash attention (wei = k q^T, causal, softmax, @v) -------
__global__ void __launch_bounds__(128)
attn_fwd(const unsigned short* __restrict__ kb, const unsigned short* __restrict__ qb,
         const unsigned short* __restrict__ vt, float* __restrict__ out) {
    __shared__ __align__(16) unsigned short pst_all[4][16 * 40]; // per-wave P staging
    int wv   = threadIdx.x >> 5;
    int lane = threadIdx.x & 31;
    int m = lane & 15, hi = lane >> 4;
    unsigned short* pst = pst_all[wv];

    int tid   = blockIdx.x * 4 + wv;     // 0..4095
    int batch = tid >> 7;
    int i0    = (tid & 127) * 16;

    const unsigned short* kbb = kb + ((long)batch * T_SEQ + i0) * HDIM;
    const unsigned short* qbb = qb + (long)batch * T_SEQ * HDIM;
    const unsigned short* vtb = vt + (long)batch * HDIM * T_SEQ;

    // k A-fragments are loop-invariant (rows i0..i0+15, h chunks 0..31 and 32..63)
    v16bf ak0 = load_a_rm(kbb + 0,  HDIM, lane);
    v16bf ak1 = load_a_rm(kbb + 32, HDIM, lane);

    const v8f vzero = {};
    v8f o[4];
    float mrow[8], lrow[8];
    #pragma unroll
    for (int h = 0; h < 4; ++h) o[h] = vzero;
    #pragma unroll
    for (int r = 0; r < 8; ++r) { mrow[r] = -1e30f; lrow[r] = 0.0f; }

    const float scale = 0.03125f;  // 1024^-0.5

    for (int j0 = 0; j0 <= i0 + 15; j0 += 32) {
        // S tile 16x32: two 16x16 f32 accumulators (j sub-tiles)
        v8f s0 = vzero, s1 = vzero;
        {
            v16bf bq00 = load_b_kc(qbb + (long)(j0)      * HDIM + 0,  HDIM, lane);
            v16bf bq01 = load_b_kc(qbb + (long)(j0)      * HDIM + 32, HDIM, lane);
            v16bf bq10 = load_b_kc(qbb + (long)(j0 + 16) * HDIM + 0,  HDIM, lane);
            v16bf bq11 = load_b_kc(qbb + (long)(j0 + 16) * HDIM + 32, HDIM, lane);
            s0 = wmma_bf16(ak0, bq00, s0);
            s0 = wmma_bf16(ak1, bq01, s0);
            s1 = wmma_bf16(ak0, bq10, s1);
            s1 = wmma_bf16(ak1, bq11, s1);
        }
        bool diag = (j0 + 31) > i0;
        #pragma unroll
        for (int r = 0; r < 8; ++r) {
            float v0 = s0[r] * scale;
            float v1 = s1[r] * scale;
            int gi = i0 + r + 8 * hi;
            if (diag) {
                if (j0 + m > gi)      v0 = -1e30f;
                if (j0 + 16 + m > gi) v1 = -1e30f;
            }
            // row reductions across the 16-lane half holding this row
            float mx = fmaxf(v0, v1);
            #pragma unroll
            for (int d = 8; d >= 1; d >>= 1) mx = fmaxf(mx, __shfl_xor(mx, d, 32));
            float mn    = fmaxf(mrow[r], mx);
            float alpha = __expf(mrow[r] - mn);
            float p0 = __expf(v0 - mn);
            float p1 = __expf(v1 - mn);
            float rs = p0 + p1;
            #pragma unroll
            for (int d = 8; d >= 1; d >>= 1) rs += __shfl_xor(rs, d, 32);
            lrow[r] = lrow[r] * alpha + rs;
            mrow[r] = mn;
            #pragma unroll
            for (int h = 0; h < 4; ++h) o[h][r] *= alpha;
            // stage P (C-layout scatter) into LDS as bf16
            pst[(r + 8 * hi) * 40 + m]      = f2bf(p0);
            pst[(r + 8 * hi) * 40 + 16 + m] = f2bf(p1);
        }
        asm volatile("s_wait_dscnt 0" ::: "memory");
        // reload P as A fragment (16x32) from LDS
        V16U pa;
        pa.h[0] = *(const u16x8*)(pst + m * 40 + 8 * hi);
        pa.h[1] = *(const u16x8*)(pst + m * 40 + 16 + 8 * hi);
        // O += P @ V  (B fragments from transposed v: two 16B runs per lane)
        #pragma unroll
        for (int h = 0; h < 4; ++h) {
            v16bf bv = load_b_kc(vtb + (long)(h * 16) * T_SEQ + j0, T_SEQ, lane);
            o[h] = wmma_bf16(pa.v, bv, o[h]);
        }
    }

    // normalize and store fp32 output [B,T,64]
    float* ob = out + ((long)batch * T_SEQ + i0) * HDIM;
    #pragma unroll
    for (int r = 0; r < 8; ++r) {
        float inv = 1.0f / lrow[r];
        #pragma unroll
        for (int h = 0; h < 4; ++h)
            ob[(long)(r + 8 * hi) * HDIM + h * 16 + m] = o[h][r] * inv;
    }
}

extern "C" void kernel_launch(void* const* d_in, const int* in_sizes, int n_in,
                              void* d_out, int out_size, void* d_ws, size_t ws_size,
                              hipStream_t stream) {
    const float* x  = (const float*)d_in[0];
    const float* Wk = (const float*)d_in[1];
    const float* Wq = (const float*)d_in[2];
    const float* Wv = (const float*)d_in[3];

    unsigned short* kb = (unsigned short*)d_ws;                      // 8 MB
    unsigned short* qb = kb + (size_t)NB * T_SEQ * HDIM;             // 8 MB
    unsigned short* vt = qb + (size_t)NB * T_SEQ * HDIM;             // 8 MB (transposed v)
    unsigned short* wt = vt + (size_t)NB * T_SEQ * HDIM;             // 384 KB (WT bf16 x3)

    // W fp32 [1024][64] -> bf16 transposed [64][1024], 3 matrices
    cvt_w<<<768, 256, 0, stream>>>(Wk, Wq, Wv, wt);
    // 4096 row-tiles, 8 waves/block
    proj_kqv<<<512, 256, 0, stream>>>(x, wt, kb, qb, vt);
    // 4096 i-tiles, 4 waves/block, one wave per tile
    attn_fwd<<<1024, 128, 0, stream>>>(kb, qb, vt, (float*)d_out);
}
